// MultiQueryAttention_30434138259532
// MI455X (gfx1250) — matrix operand order repaired and checked by reading, not compiled, not run on hardware
//
#include <hip/hip_runtime.h>
#include <hip/hip_bf16.h>

// MQA on MI455X (gfx1250, wave32): all matmuls on v_wmma_f32_16x16x32_bf16.
// Pipeline: [q@Wq] [k@Wk] [v@Wv] -> flash attention (score matrix never hits HBM)
// -> [ao@Wp + bp]. bf16 intermediates in workspace (~35.7 MB).
// bf16 tiles that need no conversion are staged memory->LDS with the CDNA5
// async path (global_load_async_to_lds_b128 + s_wait_asynccnt), bypassing VGPRs.

typedef __attribute__((ext_vector_type(16))) __bf16 v16bf;
typedef __attribute__((ext_vector_type(8)))  __bf16 bf16x8;
typedef __attribute__((ext_vector_type(8)))  float  v8f;

#define DEVINL __device__ __forceinline__

DEVINL v16bf join8(bf16x8 lo, bf16x8 hi) {
  return __builtin_shufflevector(lo, hi, 0,1,2,3,4,5,6,7,8,9,10,11,12,13,14,15);
}

DEVINL v8f wmma_bf16(v16bf a, v16bf b, v8f c) {
  // (neg_a, A, neg_b, B, c_mod, C, reuse_a, reuse_b)
  return __builtin_amdgcn_wmma_f32_16x16x32_bf16(false, a, false, b, (short)0, c, false, false);
}

// ---- CDNA5 async memory->LDS path (ASYNCcnt-tracked, no VGPR round trip) ----
// Generic pointer to a __shared__ object is {aperture_hi32, lds_offset_lo32};
// the instruction's VDST wants the LDS byte offset (HW adds LDS_BASE).
DEVINL uint32_t lds_off32(const void* p) { return (uint32_t)(uintptr_t)p; }

DEVINL void async_ld_b128(uint32_t lds_off, const void* gsrc) {
  asm volatile("global_load_async_to_lds_b128 %0, %1, off"
               :: "v"(lds_off), "v"(gsrc) : "memory");
}
DEVINL void wait_async0() {
  asm volatile("s_wait_asynccnt 0x0" ::: "memory");
}

// ---------------------------------------------------------------------------
// Generic GEMM: out[M,N] = A[M,K] @ W[K,N] (+ bias), A fp32 or bf16, out bf16
// or fp32+bias. Block tile BM x BN, K staged 64 wide. 8 waves per block; each
// wave computes WAVE_M x 64 via (WAVE_M/16) x 4 WMMA tiles.
// LDS layouts (stride 72 bf16 = 144 B: 16B aligned rows, conflict-free b128):
//   sA[row][k]  (A fragment: lane m=row, K innermost)
//   sB[n][k]    (B fragment: lane n=col, K innermost -> W staged transposed)
// ---------------------------------------------------------------------------
template<int BM, int BN, int WAVE_M, bool A_BF16, bool OUT_F32_BIAS>
__global__ __launch_bounds__(256)
void gemm_wmma_kernel(const void* __restrict__ Av, const float* __restrict__ W,
                      void* __restrict__ Outv, const float* __restrict__ bias,
                      int M, int N, int K, int lda, int ldb, int ldc)
{
  constexpr int KSTEP = 64;
  constexpr int LSTR  = KSTEP + 8;        // 72 bf16 = 144 B row stride
  constexpr int MT    = WAVE_M / 16;
  constexpr int NWM   = BM / WAVE_M;
  constexpr int NWN   = BN / 64;
  constexpr int NT    = 4;
  static_assert(NWM * NWN == 8, "need exactly 8 waves");

  __shared__ __bf16 sA[BM][LSTR];
  __shared__ __bf16 sB[BN][LSTR];

  const int tid  = threadIdx.x;
  const int wave = tid >> 5;
  const int lane = tid & 31;
  const int ln   = lane & 15;
  const int hf   = lane >> 4;

  const int wm = wave % NWM;
  const int wn = wave / NWM;

  const int rowBlock = blockIdx.x * BM;
  const int colBlock = blockIdx.y * BN;

  v8f acc[MT][NT] = {};

  for (int k0 = 0; k0 < K; k0 += KSTEP) {
    // ---- stage A: BM x KSTEP
    if constexpr (A_BF16) {
      // bf16 source: verbatim 16B blocks via async copy (ASYNCcnt)
      constexpr int ABLK = BM * (KSTEP / 8);     // 16B blocks
      for (int e = tid; e < ABLK; e += 256) {
        const int row = e / (KSTEP / 8);
        const int seg = e % (KSTEP / 8);
        async_ld_b128(lds_off32(&sA[row][seg * 8]),
                      (const __bf16*)Av + (size_t)(rowBlock + row) * lda + k0 + seg * 8);
      }
    } else {
      // f32 source: load+convert pairs, 32-bit LDS stores
      constexpr int APAIRS = BM * KSTEP / 2;
      for (int p = tid; p < APAIRS; p += 256) {
        const int row = p / (KSTEP / 2);
        const int kk  = (p % (KSTEP / 2)) * 2;
        const float* src = (const float*)Av + (size_t)(rowBlock + row) * lda + k0 + kk;
        sA[row][kk]     = (__bf16)src[0];
        sA[row][kk + 1] = (__bf16)src[1];
      }
    }
    // ---- stage B transposed: sB[n][k] = W[(k0+k)*ldb + colBlock + n]
    constexpr int BELEMS = BN * KSTEP;
    for (int e = tid; e < BELEMS; e += 256) {
      const int n  = e % BN;
      const int kk = e / BN;
      sB[n][kk] = (__bf16)W[(size_t)(k0 + kk) * ldb + colBlock + n];
    }
    // prefetch next A K-tile (global_prefetch_b8)
    if (k0 + KSTEP < K) {
      const int prow = rowBlock + (tid >> 1);
      if constexpr (A_BF16)
        __builtin_prefetch((const __bf16*)Av + (size_t)prow * lda + k0 + KSTEP);
      else
        __builtin_prefetch((const float*)Av + (size_t)prow * lda + k0 + KSTEP);
    }
    if constexpr (A_BF16) wait_async0();   // async writes visible before barrier
    __syncthreads();

    #pragma unroll
    for (int s = 0; s < KSTEP / 32; ++s) {
      v16bf afr[MT];
      #pragma unroll
      for (int mt = 0; mt < MT; ++mt) {
        // A frag: elems 0-7 <- K = s*32 + hf*8 + j ; elems 8-15 <- +16
        const __bf16* base = &sA[wm * WAVE_M + mt * 16 + ln][s * 32];
        bf16x8 lo = *(const bf16x8*)(base + hf * 8);
        bf16x8 hi = *(const bf16x8*)(base + 16 + hf * 8);
        afr[mt] = join8(lo, hi);
      }
      v16bf bfr[NT];
      #pragma unroll
      for (int t = 0; t < NT; ++t) {
        // B frag: elems 0-7 <- K = s*32 + hf*8 + j ; elems 8-15 <- +16
        const __bf16* base = &sB[wn * 64 + t * 16 + ln][s * 32];
        bf16x8 lo = *(const bf16x8*)(base + hf * 8);
        bf16x8 hi = *(const bf16x8*)(base + 16 + hf * 8);
        bfr[t] = join8(lo, hi);
      }
      #pragma unroll
      for (int mt = 0; mt < MT; ++mt)
        #pragma unroll
        for (int t = 0; t < NT; ++t)
          acc[mt][t] = wmma_bf16(afr[mt], bfr[t], acc[mt][t]);
    }
    __syncthreads();
  }

  // ---- epilogue: C layout -> VGPR r holds M = r + 8*hf, lane ln holds N
  #pragma unroll
  for (int mt = 0; mt < MT; ++mt) {
    #pragma unroll
    for (int t = 0; t < NT; ++t) {
      #pragma unroll
      for (int r = 0; r < 8; ++r) {
        const int row = rowBlock + wm * WAVE_M + mt * 16 + r + 8 * hf;
        const int col = colBlock + wn * 64 + t * 16 + ln;
        const float vv = acc[mt][t][r];
        if constexpr (OUT_F32_BIAS)
          ((float*)Outv)[(size_t)row * ldc + col] = vv + bias[col];
        else
          ((__bf16*)Outv)[(size_t)row * ldc + col] = (__bf16)vv;
      }
    }
  }
}

// ---------------------------------------------------------------------------
// Flash MQA: one block per (b, h, 128 q rows). 8 waves; each wave owns 16 q
// rows. Q kept as A-fragments in registers; K/V streamed in 64-row LDS tiles
// (K via async copy; V transposed through VGPRs). Online softmax in registers;
// P round-trips through a per-wave LDS buffer to convert C-layout ->
// A-fragment layout for the PV matmul.
// ---------------------------------------------------------------------------
__global__ __launch_bounds__(256)
void mqa_flash_kernel(const __bf16* __restrict__ qp, const __bf16* __restrict__ kp,
                      const __bf16* __restrict__ vp, __bf16* __restrict__ ao)
{
  constexpr int T = 2048, Dh = 64, CC = 1024;
  constexpr int LSTR = Dh + 8;                     // 72
  __shared__ __bf16 sK[64][LSTR];                  // [kpos][dim]  (B frag for QK^T)
  __shared__ __bf16 sV[64][LSTR];                  // [dim][kpos]  (B frag for PV)
  __shared__ __bf16 sP[8][16][LSTR];               // per-wave P relayout buffer

  const int tid  = threadIdx.x;
  const int wave = tid >> 5;
  const int lane = tid & 31;
  const int ln   = lane & 15;
  const int hf   = lane >> 4;

  const int b  = blockIdx.z;
  const int h  = blockIdx.y;
  const int qw = blockIdx.x * 128 + wave * 16;     // first q row of this wave

  // Q fragments (both 32-wide K steps of head dim), resident for whole kernel
  const __bf16* qbase = qp + ((size_t)b * T + qw + ln) * CC + h * Dh;
  v16bf aq[2];
  #pragma unroll
  for (int s = 0; s < 2; ++s) {
    bf16x8 lo = *(const bf16x8*)(qbase + s * 32 + hf * 8);
    bf16x8 hi = *(const bf16x8*)(qbase + s * 32 + 16 + hf * 8);
    aq[s] = join8(lo, hi);
  }

  float mrow[8], lrow[8];
  #pragma unroll
  for (int r = 0; r < 8; ++r) { mrow[r] = -3.0e38f; lrow[r] = 0.0f; }
  v8f oacc[4] = {};

  const float c = 0.125f * 1.44269504f;            // 1/sqrt(64) * log2(e)
  const __bf16* kpb = kp + (size_t)b * T * Dh;
  const __bf16* vpb = vp + (size_t)b * T * Dh;

  for (int kb = 0; kb < T; kb += 64) {
    __syncthreads();
    // stage K tile row-major via CDNA5 async copy (bf16 verbatim, b128 blocks)
    for (int e = tid; e < 512; e += 256) {
      const int row = e >> 3, seg = e & 7;
      async_ld_b128(lds_off32(&sK[row][seg * 8]),
                    kpb + (size_t)(kb + row) * Dh + seg * 8);
    }
    // stage V tile transposed: sV[dim][kpos] (needs lane transpose -> VGPR path)
    for (int e = tid; e < 4096; e += 256) {
      const int d = e & 63, row = e >> 6;
      sV[d][row] = vpb[(size_t)(kb + row) * Dh + d];
    }
    wait_async0();                                 // K-tile async writes done
    __syncthreads();

    // S = Q K^T : 16 q-rows x 64 k-positions per wave (8 WMMAs)
    v8f sacc[4] = {};
    #pragma unroll
    for (int s = 0; s < 2; ++s) {
      #pragma unroll
      for (int t = 0; t < 4; ++t) {
        const __bf16* base = &sK[t * 16 + ln][s * 32];
        bf16x8 lo = *(const bf16x8*)(base + hf * 8);
        bf16x8 hi = *(const bf16x8*)(base + 16 + hf * 8);
        sacc[t] = wmma_bf16(aq[s], join8(lo, hi), sacc[t]);
      }
    }

    // online softmax; VGPR r of the accumulator holds q-row (r + 8*hf),
    // so butterfly reductions with masks 1,2,4,8 stay inside each 16-lane half
    #pragma unroll
    for (int r = 0; r < 8; ++r) {
      float mx = fmaxf(fmaxf(sacc[0][r], sacc[1][r]),
                       fmaxf(sacc[2][r], sacc[3][r]));
      mx = fmaxf(mx, __shfl_xor(mx, 1, 32));
      mx = fmaxf(mx, __shfl_xor(mx, 2, 32));
      mx = fmaxf(mx, __shfl_xor(mx, 4, 32));
      mx = fmaxf(mx, __shfl_xor(mx, 8, 32));
      const float mnew  = fmaxf(mrow[r], mx);
      const float alpha = exp2f((mrow[r] - mnew) * c);
      lrow[r] *= alpha;
      #pragma unroll
      for (int t = 0; t < 4; ++t) oacc[t][r] *= alpha;
      float psum = 0.0f;
      #pragma unroll
      for (int t = 0; t < 4; ++t) {
        const float p = exp2f((sacc[t][r] - mnew) * c);
        sP[wave][r + 8 * hf][t * 16 + ln] = (__bf16)p;   // [q-row][kpos]
        psum += p;
      }
      psum += __shfl_xor(psum, 1, 32);
      psum += __shfl_xor(psum, 2, 32);
      psum += __shfl_xor(psum, 4, 32);
      psum += __shfl_xor(psum, 8, 32);
      lrow[r] += psum;
      mrow[r] = mnew;
    }

    // per-wave LDS stores must land before re-reading as A fragments
    asm volatile("s_wait_dscnt 0" ::: "memory");

    // O += P V : A = P (16 x 64 kpos), B = sV[dim][kpos] (8 WMMAs)
    #pragma unroll
    for (int s = 0; s < 2; ++s) {
      const __bf16* pb = &sP[wave][ln][s * 32];
      bf16x8 plo = *(const bf16x8*)(pb + hf * 8);
      bf16x8 phi = *(const bf16x8*)(pb + 16 + hf * 8);
      const v16bf ap = join8(plo, phi);
      #pragma unroll
      for (int t = 0; t < 4; ++t) {
        const __bf16* base = &sV[t * 16 + ln][s * 32];
        bf16x8 lo = *(const bf16x8*)(base + hf * 8);
        bf16x8 hi = *(const bf16x8*)(base + 16 + hf * 8);
        oacc[t] = wmma_bf16(ap, join8(lo, hi), oacc[t]);
      }
    }
  }

  // normalize and write attention output (bf16), row = qw + r + 8*hf
  #pragma unroll
  for (int t = 0; t < 4; ++t) {
    #pragma unroll
    for (int r = 0; r < 8; ++r) {
      const float val = oacc[t][r] / lrow[r];
      ao[((size_t)b * T + qw + r + 8 * hf) * CC + h * Dh + t * 16 + ln] = (__bf16)val;
    }
  }
}

// ---------------------------------------------------------------------------
extern "C" void kernel_launch(void* const* d_in, const int* in_sizes, int n_in,
                              void* d_out, int out_size, void* d_ws, size_t ws_size,
                              hipStream_t stream) {
  (void)in_sizes; (void)n_in; (void)out_size; (void)ws_size;
  constexpr int B = 4, T = 2048, C = 1024, Dh = 64;
  constexpr int BT = B * T;                         // 8192

  const float* q  = (const float*)d_in[0];
  const float* k  = (const float*)d_in[1];
  const float* v  = (const float*)d_in[2];
  const float* Wq = (const float*)d_in[3];
  const float* Wk = (const float*)d_in[4];
  const float* Wv = (const float*)d_in[5];
  const float* Wp = (const float*)d_in[6];
  const float* bp = (const float*)d_in[7];
  float* out = (float*)d_out;

  // workspace layout (bf16): qp[BT*C] | kp[BT*Dh] | vp[BT*Dh] | ao[BT*C]
  __bf16* qp = (__bf16*)d_ws;
  __bf16* kp = qp + (size_t)BT * C;
  __bf16* vp = kp + (size_t)BT * Dh;
  __bf16* ao = vp + (size_t)BT * Dh;

  const dim3 blk(256);

  // 1) q projection: [8192,1024] x [1024,1024] -> qp (bf16)
  gemm_wmma_kernel<128, 128, 32, false, false>
      <<<dim3(BT / 128, C / 128), blk, 0, stream>>>(q, Wq, qp, nullptr,
                                                    BT, C, C, C, C, C);
  // 2) shared-head k/v projections: [8192,1024] x [1024,64] -> bf16
  gemm_wmma_kernel<128, 64, 16, false, false>
      <<<dim3(BT / 128, 1), blk, 0, stream>>>(k, Wk, kp, nullptr,
                                              BT, Dh, C, C, Dh, Dh);
  gemm_wmma_kernel<128, 64, 16, false, false>
      <<<dim3(BT / 128, 1), blk, 0, stream>>>(v, Wv, vp, nullptr,
                                              BT, Dh, C, C, Dh, Dh);
  // 3) flash attention: grid = (q-tiles, heads, batch)
  mqa_flash_kernel<<<dim3(T / 128, 16, B), blk, 0, stream>>>(qp, kp, vp, ao);
  // 4) output projection + bias: ao(bf16) x Wp + bp -> out (f32)
  gemm_wmma_kernel<128, 128, 32, true, true>
      <<<dim3(BT / 128, C / 128), blk, 0, stream>>>(ao, Wp, out, bp,
                                                    BT, C, C, C, C, C);
}
